// SwinTransformerBlock_16741782519873
// MI455X (gfx1250) — compile-verified, matrix operand order checked by reading, not run on hardware
//
#include <hip/hip_runtime.h>
#include <hip/hip_bf16.h>
#include <math.h>

typedef __attribute__((ext_vector_type(16))) _Float16 v16h;
typedef __attribute__((ext_vector_type(8)))  float    v8f;

// Problem constants (from reference setup_inputs)
#define BB   4
#define HH   256
#define WW   256
#define CC   128
#define NH   4
#define HD   32          // head dim
#define WS   8           // window size
#define SS   4           // shift size
#define NWD  (HH / WS)   // 32 windows per dim
#define NWIN (BB * NWD * NWD)   // 4096 windows
#define NTOK 64                 // tokens per window
#define MROWS (NWIN * NTOK)     // 262144 total rows
#define HID  (4 * CC)           // 512
#define PSTR (NTOK + 8)         // padded LDS row stride for P (bank-friendly)

// ---------------------------------------------------------------------------
// WMMA fragment loader: row-major matrix, 16 rows x 32 cols tile.
// Per CDNA5 ISA 7.12.2 (16-bit A / B-operand layout):
//   lanes 0-15 : row = lane,    elems 0-7 -> K+0..7,  elems 8-15 -> K+16..23
//   lanes 16-31: row = lane-16, elems 0-7 -> K+8..15, elems 8-15 -> K+24..31
// Works for A (rows of X) and for the B operand of X@W^T (rows of W).
// ---------------------------------------------------------------------------
__device__ __forceinline__ v16h load_frag16x32(const _Float16* __restrict__ p, int ld) {
    const int lane = threadIdx.x & 31;
    const int r    = lane & 15;
    const int hi   = lane >> 4;
    const _Float16* q = p + (size_t)r * ld + hi * 8;
    v16h f;
#pragma unroll
    for (int e = 0; e < 8; ++e) f[e] = q[e];
#pragma unroll
    for (int e = 0; e < 8; ++e) f[8 + e] = q[16 + e];
    return f;
}

__device__ __forceinline__ v8f wmma_f16(v16h a, v16h b, v8f c) {
    return __builtin_amdgcn_wmma_f32_16x16x32_f16(false, a, false, b, (short)0, c, false, false);
}

__device__ __forceinline__ v8f vzero8() {
    return (v8f){0.f, 0.f, 0.f, 0.f, 0.f, 0.f, 0.f, 0.f};
}

// ---------------------------------------------------------------------------
// Kernel: f32 -> f16 conversion (weights)
// ---------------------------------------------------------------------------
__global__ void cvt_f16_kernel(const float* __restrict__ src, _Float16* __restrict__ dst, int n) {
    int i = blockIdx.x * blockDim.x + threadIdx.x;
    if (i < n) dst[i] = (_Float16)src[i];
}

// ---------------------------------------------------------------------------
// LayerNorm -> f16, one wave32 per token (128 channels => 4 per lane).
// WINMAP: destination row is in shifted-window layout (fuses roll+partition).
// ---------------------------------------------------------------------------
template <bool WINMAP>
__global__ void ln_kernel(const float* __restrict__ x,
                          const float* __restrict__ g,
                          const float* __restrict__ b,
                          _Float16* __restrict__ out) {
    const int lane = threadIdx.x & 31;
    const int wv   = threadIdx.x >> 5;
    const long long row = (long long)blockIdx.x * 8 + wv;  // dest row
    long long src;
    if (WINMAP) {
        int win = (int)(row >> 6), t = (int)(row & 63);
        int bb = win >> 10, wh = (win >> 5) & 31, ww = win & 31;
        int i = t >> 3, j = t & 7;
        int h = (wh * WS + i + SS) & (HH - 1);
        int w = (ww * WS + j + SS) & (WW - 1);
        src = (long long)bb * (HH * WW) + (long long)h * WW + w;
    } else {
        src = row;
    }
    const float4 xv = ((const float4*)(x + src * CC))[lane];
    float s  = xv.x + xv.y + xv.z + xv.w;
    float s2 = xv.x * xv.x + xv.y * xv.y + xv.z * xv.z + xv.w * xv.w;
#pragma unroll
    for (int off = 16; off >= 1; off >>= 1) {
        s  += __shfl_xor(s, off, 32);
        s2 += __shfl_xor(s2, off, 32);
    }
    const float mu = s * (1.0f / CC);
    const float rs = rsqrtf(s2 * (1.0f / CC) - mu * mu + 1e-5f);
    const int c0 = lane * 4;
    const float4 gv = ((const float4*)g)[lane];
    const float4 bv = ((const float4*)b)[lane];
    _Float16* o = out + row * CC + c0;
    o[0] = (_Float16)((xv.x - mu) * rs * gv.x + bv.x);
    o[1] = (_Float16)((xv.y - mu) * rs * gv.y + bv.y);
    o[2] = (_Float16)((xv.z - mu) * rs * gv.z + bv.z);
    o[3] = (_Float16)((xv.w - mu) * rs * gv.w + bv.w);
}

// ---------------------------------------------------------------------------
// Generic WMMA GEMM: out = X(MxK,f16) @ W^T (W is NxK,f16) + bias
// Block = 256 thr = 8 waves; block tile 256x64, wave tile 32x64
// (2 A-frags x 4 B-frags -> 8 WMMAs per k-step: 2x arithmetic intensity
//  vs 16-row waves, half the redundant B-side load issue).
// MODE 0: store f16 (ldo)            [qkv]
// MODE 1: exact GELU, store f16      [fc1]
// MODE 2: window-reverse+unshift map, out f32 = res + val  [proj]
// MODE 3: out f32 = res + val        [fc2 -> d_out]
// ---------------------------------------------------------------------------
template <int MODE>
__global__ void gemm_wmma_kernel(const _Float16* __restrict__ A,
                                 const _Float16* __restrict__ Wt,  // N x K row-major
                                 const float* __restrict__ bias,
                                 int K, int ldo,
                                 _Float16* __restrict__ outH,
                                 float* __restrict__ outF,
                                 const float* __restrict__ res) {
    const int wave = threadIdx.x >> 5;
    const int lane = threadIdx.x & 31;
    const int hi   = lane >> 4;
    const int nl   = lane & 15;
    const long long m0 = (long long)blockIdx.x * 256 + wave * 32;
    const int n0 = blockIdx.y * 64;

    v8f acc[2][4];
#pragma unroll
    for (int mi = 0; mi < 2; ++mi)
#pragma unroll
        for (int nt = 0; nt < 4; ++nt) acc[mi][nt] = vzero8();

#pragma unroll 2
    for (int k0 = 0; k0 < K; k0 += 32) {
        // pull next K-slice of the streaming A operand toward the WGP
        __builtin_prefetch(A + (m0 + (lane & 15)) * K + k0 + 32, 0, 3);
        v16h a0 = load_frag16x32(A + m0 * K + k0, K);
        v16h a1 = load_frag16x32(A + (m0 + 16) * K + k0, K);
#pragma unroll
        for (int nt = 0; nt < 4; ++nt) {
            v16h b = load_frag16x32(Wt + (size_t)(n0 + nt * 16) * K + k0, K);
            acc[0][nt] = wmma_f16(a0, b, acc[0][nt]);
            acc[1][nt] = wmma_f16(a1, b, acc[1][nt]);
        }
    }

#pragma unroll
    for (int mi = 0; mi < 2; ++mi)
#pragma unroll
    for (int nt = 0; nt < 4; ++nt) {
        const int n = n0 + nt * 16 + nl;
        const float bv = bias[n];
#pragma unroll
        for (int v = 0; v < 8; ++v) {
            const long long m = m0 + mi * 16 + v + 8 * hi;
            float val = acc[mi][nt][v] + bv;
            if (MODE == 0) {
                outH[m * ldo + n] = (_Float16)val;
            } else if (MODE == 1) {
                val = 0.5f * val * (1.0f + erff(val * 0.70710678118654752f));
                outH[m * ldo + n] = (_Float16)val;
            } else if (MODE == 2) {
                // row m is (win,token); map back through window-reverse + roll(+SS)
                int win = (int)(m >> 6), t = (int)(m & 63);
                int bb = win >> 10, wh = (win >> 5) & 31, ww = win & 31;
                int i = t >> 3, j = t & 7;
                int h = (wh * WS + i + SS) & (HH - 1);
                int w = (ww * WS + j + SS) & (WW - 1);
                long long tok = (long long)bb * (HH * WW) + (long long)h * WW + w;
                outF[tok * CC + n] = res[tok * CC + n] + val;
            } else {  // MODE 3
                outF[m * CC + n] = res[m * CC + n] + val;
            }
        }
    }
}

// ---------------------------------------------------------------------------
// Windowed attention: 1 block (128 thr = 4 waves) per window, 1 wave = 1 head.
// S = scale*(Q K^T) + rel-pos bias + shift mask ; softmax ; O = P V.
// Processed one 16-row M-tile at a time to keep <=4 f32 accumulators live
// (no VGPR spills).  V staged in LDS via one coalesced copy; P staged in LDS
// (padded stride 72 -> the 16 row starts map to distinct bank groups).
// ---------------------------------------------------------------------------
__device__ __forceinline__ int region3(int g) { return g < (HH - WS) ? 0 : (g < (HH - SS) ? 1 : 2); }

__global__ void attn_kernel(const _Float16* __restrict__ qkv,   // MROWS x 384
                            const float* __restrict__ rpb,      // 225 x 4
                            _Float16* __restrict__ ctx) {       // MROWS x 128
    __shared__ float     s_rpb[(2 * WS - 1) * (2 * WS - 1) * NH];   // 3.6 KB
    __shared__ _Float16  s_V[NTOK * CC];                            // 16 KB
    __shared__ _Float16  s_P[NH][NTOK * PSTR];                      // 36 KB

    const int win  = blockIdx.x;
    const int head = threadIdx.x >> 5;
    const int lane = threadIdx.x & 31;
    const int hi   = lane >> 4;
    const int nl   = lane & 15;
    const int wh   = (win >> 5) & 31;
    const int ww   = win & 31;

    const _Float16* base = qkv + (size_t)win * NTOK * (3 * CC);
    const int qoff = head * HD;

    // cooperative loads: rel-pos bias table + whole V block (coalesced b128)
    for (int i = threadIdx.x; i < (2 * WS - 1) * (2 * WS - 1) * NH; i += blockDim.x)
        s_rpb[i] = rpb[i];
    for (int i = threadIdx.x; i < NTOK * CC / 8; i += blockDim.x) {
        const int rowt = i >> 4;          // token
        const int grp  = i & 15;          // 8-half group within 128 channels
        ((float4*)s_V)[i] = *(const float4*)(base + (size_t)rowt * (3 * CC) + 2 * CC + grp * 8);
    }
    __syncthreads();

    // ---- K fragments (persist across M-tiles): 4 x v16h ------------------
    v16h kf[4];
#pragma unroll
    for (int nt = 0; nt < 4; ++nt)
        kf[nt] = load_frag16x32(base + (size_t)(nt * 16) * (3 * CC) + CC + qoff, 3 * CC);

    const float scale = 0.17677669529663688f;  // 1/sqrt(32)

    // ---- per-M-tile: S row-block -> softmax -> P into LDS ----------------
#pragma unroll
    for (int mt = 0; mt < 4; ++mt) {
        v16h qf = load_frag16x32(base + (size_t)(mt * 16) * (3 * CC) + qoff, 3 * CC);
        v8f S[4];
#pragma unroll
        for (int nt = 0; nt < 4; ++nt) {
            S[nt] = vzero8();
            S[nt] = wmma_f16(qf, kf[nt], S[nt]);
        }
#pragma unroll
        for (int v = 0; v < 8; ++v) {
            const int m  = mt * 16 + v + 8 * hi;   // query token
            const int i1 = m >> 3, j1 = m & 7;
            const int r1 = region3(wh * WS + i1) * 3 + region3(ww * WS + j1);
            float mx = -3.0e38f;
#pragma unroll
            for (int nt = 0; nt < 4; ++nt) {
                const int n  = nt * 16 + nl;       // key token
                const int i2 = n >> 3, j2 = n & 7;
                const int r2 = region3(wh * WS + i2) * 3 + region3(ww * WS + j2);
                const float bsv = s_rpb[((i1 - i2 + WS - 1) * (2 * WS - 1) + (j1 - j2 + WS - 1)) * NH + head];
                float sv = S[nt][v] * scale + bsv;
                if (r1 != r2) sv -= 100.0f;
                S[nt][v] = sv;
                mx = fmaxf(mx, sv);
            }
#pragma unroll
            for (int off = 8; off >= 1; off >>= 1) mx = fmaxf(mx, __shfl_xor(mx, off, 32));
            float sm = 0.f;
#pragma unroll
            for (int nt = 0; nt < 4; ++nt) {
                float e = __expf(S[nt][v] - mx);
                S[nt][v] = e;
                sm += e;
            }
#pragma unroll
            for (int off = 8; off >= 1; off >>= 1) sm += __shfl_xor(sm, off, 32);
            const float inv = 1.0f / sm;
#pragma unroll
            for (int nt = 0; nt < 4; ++nt)
                s_P[head][m * PSTR + nt * 16 + nl] = (_Float16)(S[nt][v] * inv);
        }
    }
    // per-wave LDS region + same-wave DS ordering: no cross-wave sync needed

    // ---- B fragments of V from LDS (K = token dim, N = head channels) ----
    v16h vf[2][2];
#pragma unroll
    for (int kt = 0; kt < 2; ++kt)
#pragma unroll
        for (int nt = 0; nt < 2; ++nt) {
            const int c = qoff + nt * 16 + nl;
            v16h f;
#pragma unroll
            for (int e = 0; e < 8; ++e)
                f[e] = s_V[(kt * 32 + hi * 8 + e) * CC + c];
#pragma unroll
            for (int e = 0; e < 8; ++e)
                f[8 + e] = s_V[(kt * 32 + 16 + hi * 8 + e) * CC + c];
            vf[kt][nt] = f;
        }

    // ---- O = P V, one M-tile at a time ------------------------------------
#pragma unroll
    for (int mt = 0; mt < 4; ++mt) {
        v8f O[2];
        O[0] = vzero8();
        O[1] = vzero8();
#pragma unroll
        for (int kt = 0; kt < 2; ++kt) {
            v16h pf = load_frag16x32(&s_P[head][(mt * 16) * PSTR + kt * 32], PSTR);
#pragma unroll
            for (int nt = 0; nt < 2; ++nt)
                O[nt] = wmma_f16(pf, vf[kt][nt], O[nt]);
        }
#pragma unroll
        for (int nt = 0; nt < 2; ++nt)
#pragma unroll
            for (int v = 0; v < 8; ++v) {
                const int m = mt * 16 + v + 8 * hi;
                const int c = qoff + nt * 16 + nl;
                ctx[((size_t)win * NTOK + m) * CC + c] = (_Float16)O[nt][v];
            }
    }
}

// ---------------------------------------------------------------------------
// Host-side orchestration
// ---------------------------------------------------------------------------
extern "C" void kernel_launch(void* const* d_in, const int* in_sizes, int n_in,
                              void* d_out, int out_size, void* d_ws, size_t ws_size,
                              hipStream_t stream) {
    const float* x       = (const float*)d_in[0];
    const float* n1g     = (const float*)d_in[1];
    const float* n1b     = (const float*)d_in[2];
    const float* qkv_w   = (const float*)d_in[3];
    const float* qkv_b   = (const float*)d_in[4];
    const float* rpb     = (const float*)d_in[5];
    const float* proj_w  = (const float*)d_in[6];
    const float* proj_b  = (const float*)d_in[7];
    const float* n2g     = (const float*)d_in[8];
    const float* n2b     = (const float*)d_in[9];
    const float* fc1_w   = (const float*)d_in[10];
    const float* fc1_b   = (const float*)d_in[11];
    const float* fc2_w   = (const float*)d_in[12];
    const float* fc2_b   = (const float*)d_in[13];
    float* out           = (float*)d_out;

    // workspace carve-out
    size_t off = 0;
    auto carve = [&](size_t bytes) -> void* {
        void* p = (char*)d_ws + off;
        off += (bytes + 255) & ~(size_t)255;
        return p;
    };
    _Float16* w_qkv  = (_Float16*)carve((size_t)3 * CC * CC * 2);
    _Float16* w_proj = (_Float16*)carve((size_t)CC * CC * 2);
    _Float16* w_fc1  = (_Float16*)carve((size_t)HID * CC * 2);
    _Float16* w_fc2  = (_Float16*)carve((size_t)CC * HID * 2);
    _Float16* win16  = (_Float16*)carve((size_t)MROWS * CC * 2);
    _Float16* qkv16  = (_Float16*)carve((size_t)MROWS * 3 * CC * 2);
    _Float16* ctx16  = (_Float16*)carve((size_t)MROWS * CC * 2);
    float*    x1     = (float*)   carve((size_t)MROWS * CC * 4);
    _Float16* y16    = (_Float16*)carve((size_t)MROWS * CC * 2);
    _Float16* h16    = (_Float16*)carve((size_t)MROWS * HID * 2);
    (void)ws_size; (void)n_in; (void)in_sizes; (void)out_size;

    // 1) convert weights to f16
    cvt_f16_kernel<<<(3*CC*CC + 255)/256, 256, 0, stream>>>(qkv_w,  w_qkv,  3*CC*CC);
    cvt_f16_kernel<<<(CC*CC   + 255)/256, 256, 0, stream>>>(proj_w, w_proj, CC*CC);
    cvt_f16_kernel<<<(HID*CC  + 255)/256, 256, 0, stream>>>(fc1_w,  w_fc1,  HID*CC);
    cvt_f16_kernel<<<(CC*HID  + 255)/256, 256, 0, stream>>>(fc2_w,  w_fc2,  CC*HID);

    // 2) LN1 fused with roll(-4,-4) + window partition -> f16
    ln_kernel<true><<<MROWS / 8, 256, 0, stream>>>(x, n1g, n1b, win16);

    // 3) QKV GEMM: 262144 x 384 x 128
    gemm_wmma_kernel<0><<<dim3(MROWS/256, (3*CC)/64), 256, 0, stream>>>(
        win16, w_qkv, qkv_b, CC, 3*CC, qkv16, nullptr, nullptr);

    // 4) windowed attention (WMMA QK^T + softmax + WMMA PV)
    attn_kernel<<<NWIN, 128, 0, stream>>>(qkv16, rpb, ctx16);

    // 5) proj GEMM + window-reverse + roll(+4,+4) + residual -> x1 (f32)
    gemm_wmma_kernel<2><<<dim3(MROWS/256, CC/64), 256, 0, stream>>>(
        ctx16, w_proj, proj_b, CC, CC, nullptr, x1, x);

    // 6) LN2 -> f16
    ln_kernel<false><<<MROWS / 8, 256, 0, stream>>>(x1, n2g, n2b, y16);

    // 7) FC1 GEMM + exact GELU: 262144 x 512 x 128
    gemm_wmma_kernel<1><<<dim3(MROWS/256, HID/64), 256, 0, stream>>>(
        y16, w_fc1, fc1_b, CC, HID, h16, nullptr, nullptr);

    // 8) FC2 GEMM + residual -> d_out: 262144 x 128 x 512
    gemm_wmma_kernel<3><<<dim3(MROWS/256, CC/64), 256, 0, stream>>>(
        h16, w_fc2, fc2_b, HID, CC, nullptr, out, x1);
}
